// InfoNCE_17884243820878
// MI455X (gfx1250) — matrix operand order repaired and checked by reading, not compiled, not run on hardware
//
#include <hip/hip_runtime.h>
#include <hip/hip_bf16.h>
#include <math.h>

constexpr int NB = 128;   // batch (both GEMM dims)
constexpr int NT = 512;   // time
constexpr int ND = 1024;  // feature
constexpr int NKC = ND / 32;  // K-chunks of 32

typedef __attribute__((ext_vector_type(16))) __bf16 v16bf;
typedef __attribute__((ext_vector_type(8)))  __bf16 v8bf;
typedef __attribute__((ext_vector_type(8)))  float  v8f;

// LDS row stride in bf16 elements: 32 data + 8 pad = 40 (80B: 16B-aligned rows,
// 20-bank stride -> conflict-free strided b128 reads)
constexpr int LDSTR = 40;

__device__ __forceinline__ void split_bf(float v, __bf16& hi, __bf16& lo) {
  unsigned u = __float_as_uint(v) & 0xFFFF0000u;
  float hf = __uint_as_float(u);   // exactly representable in bf16
  hi = (__bf16)hf;
  lo = (__bf16)(v - hf);           // residual, RNE to bf16
}

__device__ __forceinline__ v16bf cat16(v8bf a, v8bf b) {
  return __builtin_shufflevector(a, b, 0,1,2,3,4,5,6,7,8,9,10,11,12,13,14,15);
}

// ---------------------------------------------------------------------------
// Kernel 1: per-t 128x128xK GEMM via bf16x3 WMMA, double-buffered LDS with
// register prefetch of the next K-chunk. logits layout: [t][i][j]
// ---------------------------------------------------------------------------
__global__ __launch_bounds__(256) void gemm_logits(const float* __restrict__ x,
                                                   const float* __restrict__ c,
                                                   float* __restrict__ logits) {
  __shared__ __bf16 lxh[2][NB * LDSTR];
  __shared__ __bf16 lxl[2][NB * LDSTR];
  __shared__ __bf16 lch[2][NB * LDSTR];
  __shared__ __bf16 lcl[2][NB * LDSTR];

  const int t    = blockIdx.x;
  const int tid  = threadIdx.x;
  const int lane = tid & 31;
  const int wv   = tid >> 5;       // wave id 0..7 -> row strip
  const int m    = lane & 15;
  const int half = lane >> 4;

  v8f acc[8];
  #pragma unroll
  for (int n = 0; n < 8; ++n)
    acc[n] = (v8f){0.f,0.f,0.f,0.f,0.f,0.f,0.f,0.f};

  // global loader mapping: 2 threads per row, 16 consecutive floats each
  const int gr = tid >> 1;          // 0..127
  const int gk = (tid & 1) * 16;    // 0 or 16
  const float* xrow = x + ((size_t)gr * NT + t) * ND + gk;
  const float* crow = c + ((size_t)gr * NT + t) * ND + gk;
  const int soff = gr * LDSTR + gk;

  // A/B read offsets (contiguous k order k = 16*half + s, same for A and B)
  const int arow = wv * 16 + m;
  const int ka0  = half * 16;        // A: lane half holds 16 consecutive K (first 8)
  const int ka1  = half * 16 + 8;    // A: next 8 consecutive K
  const int kb   = half * 16;        // B: lane half holds 16 consecutive K

  // prefetch registers for the next K-chunk
  float rx[16], rc[16];

  // ---- prologue: fetch chunk 0, stage into buffer 0 ----
  #pragma unroll
  for (int q = 0; q < 4; ++q) {
    float4 vx = *(const float4*)(xrow + 4 * q);
    float4 vc = *(const float4*)(crow + 4 * q);
    rx[4*q+0] = vx.x; rx[4*q+1] = vx.y; rx[4*q+2] = vx.z; rx[4*q+3] = vx.w;
    rc[4*q+0] = vc.x; rc[4*q+1] = vc.y; rc[4*q+2] = vc.z; rc[4*q+3] = vc.w;
  }
  {
    v8bf xh0, xh1, xl0, xl1, ch0, ch1, cl0, cl1;
    #pragma unroll
    for (int e = 0; e < 8; ++e) {
      __bf16 h, l;
      split_bf(rx[e],     h, l); xh0[e] = h; xl0[e] = l;
      split_bf(rx[e + 8], h, l); xh1[e] = h; xl1[e] = l;
      split_bf(rc[e],     h, l); ch0[e] = h; cl0[e] = l;
      split_bf(rc[e + 8], h, l); ch1[e] = h; cl1[e] = l;
    }
    *(v8bf*)&lxh[0][soff]     = xh0;  *(v8bf*)&lxh[0][soff + 8] = xh1;
    *(v8bf*)&lxl[0][soff]     = xl0;  *(v8bf*)&lxl[0][soff + 8] = xl1;
    *(v8bf*)&lch[0][soff]     = ch0;  *(v8bf*)&lch[0][soff + 8] = ch1;
    *(v8bf*)&lcl[0][soff]     = cl0;  *(v8bf*)&lcl[0][soff + 8] = cl1;
  }
  __syncthreads();

  #pragma unroll 1
  for (int kc = 0; kc < NKC; ++kc) {
    const int cur = kc & 1;
    const int nxt = cur ^ 1;

    // ---- issue global loads for chunk kc+1 (latency hidden behind WMMAs) ----
    if (kc + 1 < NKC) {
      const float* xp = xrow + (kc + 1) * 32;
      const float* cp = crow + (kc + 1) * 32;
      #pragma unroll
      for (int q = 0; q < 4; ++q) {
        float4 vx = *(const float4*)(xp + 4 * q);
        float4 vc = *(const float4*)(cp + 4 * q);
        rx[4*q+0] = vx.x; rx[4*q+1] = vx.y; rx[4*q+2] = vx.z; rx[4*q+3] = vx.w;
        rc[4*q+0] = vc.x; rc[4*q+1] = vc.y; rc[4*q+2] = vc.z; rc[4*q+3] = vc.w;
      }
    }

    // ---- compute from buf[cur] ----
    const __bf16* bxh = &lxh[cur][0];
    const __bf16* bxl = &lxl[cur][0];
    const __bf16* bch = &lch[cur][0];
    const __bf16* bcl = &lcl[cur][0];

    v16bf a_hi = cat16(*(const v8bf*)&bxh[arow * LDSTR + ka0],
                       *(const v8bf*)&bxh[arow * LDSTR + ka1]);
    v16bf a_lo = cat16(*(const v8bf*)&bxl[arow * LDSTR + ka0],
                       *(const v8bf*)&bxl[arow * LDSTR + ka1]);

    #pragma unroll
    for (int n = 0; n < 8; ++n) {
      const int bcol = n * 16 + m;
      v16bf b_hi = cat16(*(const v8bf*)&bch[bcol * LDSTR + kb],
                         *(const v8bf*)&bch[bcol * LDSTR + kb + 8]);
      v16bf b_lo = cat16(*(const v8bf*)&bcl[bcol * LDSTR + kb],
                         *(const v8bf*)&bcl[bcol * LDSTR + kb + 8]);
      acc[n] = __builtin_amdgcn_wmma_f32_16x16x32_bf16(
                   false, a_hi, false, b_hi, (short)0, acc[n], false, false);
      acc[n] = __builtin_amdgcn_wmma_f32_16x16x32_bf16(
                   false, a_hi, false, b_lo, (short)0, acc[n], false, false);
      acc[n] = __builtin_amdgcn_wmma_f32_16x16x32_bf16(
                   false, a_lo, false, b_hi, (short)0, acc[n], false, false);
    }

    // ---- stage prefetched chunk into buf[nxt] (last read of nxt was kc-1,
    //      fenced by the previous barrier) ----
    if (kc + 1 < NKC) {
      v8bf xh0, xh1, xl0, xl1, ch0, ch1, cl0, cl1;
      #pragma unroll
      for (int e = 0; e < 8; ++e) {
        __bf16 h, l;
        split_bf(rx[e],     h, l); xh0[e] = h; xl0[e] = l;
        split_bf(rx[e + 8], h, l); xh1[e] = h; xl1[e] = l;
        split_bf(rc[e],     h, l); ch0[e] = h; cl0[e] = l;
        split_bf(rc[e + 8], h, l); ch1[e] = h; cl1[e] = l;
      }
      *(v8bf*)&lxh[nxt][soff]     = xh0;  *(v8bf*)&lxh[nxt][soff + 8] = xh1;
      *(v8bf*)&lxl[nxt][soff]     = xl0;  *(v8bf*)&lxl[nxt][soff + 8] = xl1;
      *(v8bf*)&lch[nxt][soff]     = ch0;  *(v8bf*)&lch[nxt][soff + 8] = ch1;
      *(v8bf*)&lcl[nxt][soff]     = cl0;  *(v8bf*)&lcl[nxt][soff + 8] = cl1;
      __syncthreads();   // single barrier per iteration
    }
  }

  // ---- epilogue: C layout (VGPR r -> row r + 8*half) -> logits[t][i][j] ----
  float* lt = logits + (size_t)t * NB * NB;
  #pragma unroll
  for (int n = 0; n < 8; ++n) {
    #pragma unroll
    for (int r = 0; r < 8; ++r) {
      const int row = wv * 16 + half * 8 + r;
      const int col = n * 16 + m;
      lt[row * NB + col] = acc[n][r];
    }
  }
}

// ---------------------------------------------------------------------------
// Kernel 2: log_softmax over j (axis=1) + diagonal-loss partials.
// grid: (NB, NT/32). Block handles row i for 32 consecutive t.
// ---------------------------------------------------------------------------
__global__ __launch_bounds__(256) void softmax_rows(const float* __restrict__ logits,
                                                    float* __restrict__ ls1,
                                                    float* __restrict__ partials) {
  __shared__ float tile[32 * 129];
  __shared__ float lse[32];
  __shared__ float dred[32];
  const int tid = threadIdx.x;
  const int i   = blockIdx.x;
  const int t0  = blockIdx.y * 32;

  // load tile[tt][j] = logits[t0+tt][i][j]   (coalesced 64B per thread)
  {
    const int tt = tid >> 3;
    const int jb = (tid & 7) * 16;
    const float* src = logits + ((size_t)(t0 + tt) * NB + i) * NB + jb;
    #pragma unroll
    for (int q = 0; q < 4; ++q) {
      float4 v = *(const float4*)(src + q * 4);
      float* d = &tile[tt * 129 + jb + q * 4];
      d[0] = v.x; d[1] = v.y; d[2] = v.z; d[3] = v.w;
    }
  }
  __syncthreads();

  if (tid < 32) {
    const float* row = &tile[tid * 129];
    float mx = row[0];
    for (int j = 1; j < NB; ++j) mx = fmaxf(mx, row[j]);
    float s = 0.f;
    for (int j = 0; j < NB; ++j) s += expf(row[j] - mx);
    const float l = mx + logf(s);
    lse[tid]  = l;
    dred[tid] = row[i] - l;           // diagonal ls1 value for (i,i,t0+tid)
  }
  __syncthreads();

  // write ls1[i][j][t0+q]: consecutive tid -> consecutive t (coalesced runs)
  for (int idx = tid; idx < 32 * NB; idx += 256) {
    const int j = idx >> 5, q = idx & 31;
    ls1[(size_t)i * NB * NT + (size_t)j * NT + t0 + q] = tile[q * 129 + j] - lse[q];
  }
  if (tid == 0) {
    float s = 0.f;
    for (int q = 0; q < 32; ++q) s += dred[q];   // fixed order
    partials[(size_t)blockIdx.y * NB + i] = s;
  }
}

// ---------------------------------------------------------------------------
// Kernel 3: log_softmax over i (axis=0). Block handles column j for 32 t.
// ---------------------------------------------------------------------------
__global__ __launch_bounds__(256) void softmax_cols(const float* __restrict__ logits,
                                                    float* __restrict__ ls2) {
  __shared__ float tile[32 * 129];
  __shared__ float lse[32];
  const int tid = threadIdx.x;
  const int j   = blockIdx.x;
  const int t0  = blockIdx.y * 32;

  // tile[tt][i] = logits[t0+tt][i][j]  (strided reads; logits is L2-resident)
  {
    const int tt = tid >> 3;
    const int ib = (tid & 7) * 16;
    const float* src = logits + (size_t)(t0 + tt) * NB * NB + j;
    #pragma unroll
    for (int e = 0; e < 16; ++e)
      tile[tt * 129 + ib + e] = src[(size_t)(ib + e) * NB];
  }
  __syncthreads();

  if (tid < 32) {
    const float* row = &tile[tid * 129];
    float mx = row[0];
    for (int ii = 1; ii < NB; ++ii) mx = fmaxf(mx, row[ii]);
    float s = 0.f;
    for (int ii = 0; ii < NB; ++ii) s += expf(row[ii] - mx);
    lse[tid] = mx + logf(s);
  }
  __syncthreads();

  for (int idx = tid; idx < 32 * NB; idx += 256) {
    const int ii = idx >> 5, q = idx & 31;
    ls2[(size_t)ii * NB * NT + (size_t)j * NT + t0 + q] = tile[q * 129 + ii] - lse[q];
  }
}

// ---------------------------------------------------------------------------
// Kernel 4: deterministic fixed-order reduction of diagonal partials -> loss
// ---------------------------------------------------------------------------
__global__ __launch_bounds__(256) void reduce_loss(const float* __restrict__ partials,
                                                   float* __restrict__ out) {
  __shared__ float red[256];
  float s = 0.f;
  for (int idx = threadIdx.x; idx < (NT / 32) * NB; idx += 256) s += partials[idx];
  red[threadIdx.x] = s;
  __syncthreads();
  for (int st = 128; st > 0; st >>= 1) {
    if (threadIdx.x < st) red[threadIdx.x] += red[threadIdx.x + st];
    __syncthreads();
  }
  if (threadIdx.x == 0)
    out[0] = -0.5f * red[0] / (float)((size_t)NB * NT);
}

// ---------------------------------------------------------------------------
extern "C" void kernel_launch(void* const* d_in, const int* in_sizes, int n_in,
                              void* d_out, int out_size, void* d_ws, size_t ws_size,
                              hipStream_t stream) {
  (void)in_sizes; (void)n_in; (void)out_size; (void)ws_size;
  const float* x = (const float*)d_in[0];
  const float* c = (const float*)d_in[1];
  float* out      = (float*)d_out;
  float* logits   = (float*)d_ws;                         // 512*128*128 f32 = 33.5MB
  float* partials = logits + (size_t)NB * NB * NT;        // 2048 f32
  float* ls1 = out + 1;
  float* ls2 = out + 1 + (size_t)NB * NB * NT;

  gemm_logits<<<NT, 256, 0, stream>>>(x, c, logits);
  dim3 g(NB, NT / 32);
  softmax_rows<<<g, 256, 0, stream>>>(logits, ls1, partials);
  softmax_cols<<<g, 256, 0, stream>>>(logits, ls2);
  reduce_loss<<<1, 256, 0, stream>>>(partials, out);
}
